// ScaledDotProductAttention_26877905338507
// MI455X (gfx1250) — matrix operand order repaired and checked by reading, not compiled, not run on hardware
//
#include <hip/hip_runtime.h>
#include <hip/hip_bf16.h>

// ---------------------------------------------------------------------------
// Music-Transformer relative attention, MI455X (gfx1250, wave32, WMMA).
//
// attn[i,j] = softmax_j( (q_i.k_j + q_i.e_{j+L-1-i}) / 8 ) with causal mask,
// out = attn @ v.  Outputs concatenated: [B,L,D] output then [B,L,L] attn.
//
// Design: one workgroup per (batch, 16-row stripe). The full 16x2048 fp32
// logit stripe lives in LDS (of CDNA5's 320KB/WGP), so the 268MB attn
// matrix is written to HBM exactly once (non-temporal) -> ~12us BW floor.
// All three matmuls use v_wmma_f32_16x16x32_f16 (f16 in, f32 accumulate).
// K / E-band / V tiles are staged global->LDS with CDNA5 async data movers
// (global_load_async_to_lds_b128, drained via s_wait_asynccnt) -- no VGPR
// round trip, no loadcnt stalls; fp32->f16 happens in the fragment builders.
// The relative-position skew is computed per tile as a 16x32 band GEMM
// T = Q @ E_band^T followed by a diagonal gather T[r, c-r+15] via LDS.
// ---------------------------------------------------------------------------

typedef _Float16 v16h __attribute__((ext_vector_type(16)));
typedef _Float16 v8h  __attribute__((ext_vector_type(8)));
typedef float    v8f  __attribute__((ext_vector_type(8)));
typedef float    v4f  __attribute__((ext_vector_type(4)));   // native vec for NT stores

#define WMMA_F16(a, b, c) \
  __builtin_amdgcn_wmma_f32_16x16x32_f16(false, (a), false, (b), (short)0, (c), false, false)

constexpr int B_SZ  = 16;
constexpr int L_SZ  = 2048;
constexpr int D_SZ  = 64;
constexpr int TILES = L_SZ / 16;           // 128 row-tiles per batch
constexpr float TEMP_INV = 0.125f;         // 1 / sqrt(d_k) = 1/8
constexpr float NEG_BIG  = -1.0e30f;

// ---- dynamic LDS layout (bytes) -------------------------------------------
constexpr int SM_P   = 0;                          // float [16][2048] logits stripe
constexpr int SM_Q   = SM_P   + 16 * L_SZ * 4;     // f16   [16][64]   Q stripe
constexpr int SM_KST = SM_Q   + 16 * 64 * 2;       // float [8][16*64] per-wave K stage (async dst)
constexpr int SM_EST = SM_KST + 8 * 16 * 64 * 4;   // float [8][32*64] per-wave E-band / V stage (async dst)
constexpr int SM_TS  = SM_EST + 8 * 32 * 64 * 4;   // float [8][16*32] per-wave T scratch
constexpr int SM_RED = SM_TS  + 8 * 16 * 32 * 4;   // float [16][16]   row reductions
constexpr int SM_M   = SM_RED + 16 * 16 * 4;       // float [16]       row max
constexpr int SM_DI  = SM_M   + 16 * 4;            // float [16]       1/row-sum
constexpr int SMEM_BYTES = SM_DI + 16 * 4;         // = 248960 (< 320KB/WGP)

// ---- CDNA5 async global->LDS copy (16B per lane, ASYNCcnt-tracked) --------
__device__ __forceinline__ void async_b128(unsigned lds_byte_off, const float* gsrc) {
  asm volatile("global_load_async_to_lds_b128 %0, %1, off"
               :: "v"(lds_byte_off), "v"((unsigned long long)(const void*)gsrc)
               : "memory");
}
__device__ __forceinline__ void wait_async0() {
  asm volatile("s_wait_asynccnt 0x0" ::: "memory");
}
__device__ __forceinline__ unsigned lds_off(const void* p) {
  // generic LDS-aperture address keeps the LDS byte offset in the low 32 bits
  return (unsigned)(unsigned long long)p;
}

// ---- WMMA operand loaders (layouts per CDNA5 ISA 7.12.2) ------------------
// A (16x32 f16): lane: M = lane&15, half h = lane>>4.
//   elements 0..7  -> K = kk + 8h + i ; elements 8..15 -> K = kk + 16 + 8h + i
__device__ __forceinline__ v16h load_a16(const _Float16* mat, int m, int kk, int h) {
  const _Float16* p = mat + m * 64 + kk + h * 8;
  v8h lo = *(const v8h*)p;
  v8h hi = *(const v8h*)(p + 16);
  v16h a;
#pragma unroll
  for (int i = 0; i < 8; ++i) { a[i] = lo[i]; a[i + 8] = hi[i]; }
  return a;
}

// B (32x16 f16 from fp32 row-major stage [N][64]): N = lane&15, K = 8h + e (e<8), 16 + 8h + (e-8) (e>=8)
__device__ __forceinline__ v16h load_b_f32row(const float* mat, int n, int kk, int h) {
  const float* p = mat + n * 64 + kk + h * 8;
  v16h bv;
#pragma unroll
  for (int i = 0; i < 16; ++i) bv[i] = (_Float16)p[i < 8 ? i : i + 8];
  return bv;
}

// A from fp32 probability stripe (convert to f16 on the fly), K block = 32 cols at jb
__device__ __forceinline__ v16h load_a_f32(const float* P, int m, int jb, int h) {
  const float* p0 = P + m * L_SZ + jb + h * 8;
  const float* p1 = p0 + 16;
  v16h a;
#pragma unroll
  for (int i = 0; i < 8; ++i) { a[i] = (_Float16)p0[i]; a[i + 8] = (_Float16)p1[i]; }
  return a;
}

// B for PV from V staged fp32 row-major [32 j][64 d]: N = d, K = j-offset = 8h + e (e<8), 16 + 8h + (e-8) (e>=8)
__device__ __forceinline__ v16h load_b_vcol(const float* Vst, int dcol, int h) {
  v16h bv;
#pragma unroll
  for (int i = 0; i < 16; ++i) bv[i] = (_Float16)Vst[(h * 8 + (i < 8 ? i : i + 8)) * 64 + dcol];
  return bv;
}

__global__ __launch_bounds__(256, 1)
void relattn_fused_kernel(const float* __restrict__ qg, const float* __restrict__ kg,
                          const float* __restrict__ vg, const float* __restrict__ eg,
                          float* __restrict__ out) {
  extern __shared__ __align__(32) char smem[];
  float*    Plds = (float*)(smem + SM_P);
  _Float16* Qs   = (_Float16*)(smem + SM_Q);
  float*    red  = (float*)(smem + SM_RED);
  float*    mrow = (float*)(smem + SM_M);
  float*    dinv = (float*)(smem + SM_DI);

  const int tid  = threadIdx.x;
  const int w    = tid >> 5;        // wave id 0..7
  const int lane = tid & 31;
  const int h    = lane >> 4;       // lane half
  const int ml   = lane & 15;

  float* Kst = (float*)(smem + SM_KST) + w * (16 * 64);
  float* Est = (float*)(smem + SM_EST) + w * (32 * 64);
  float* tsc = (float*)(smem + SM_TS)  + w * (16 * 32);
  const unsigned kst_off = lds_off(Kst);
  const unsigned est_off = lds_off(Est);

  const int bid = blockIdx.x;
  const int b   = bid >> 7;         // batch
  const int it  = bid & 127;        // row-tile index
  const int i0  = it << 4;
  const int ncols   = (it + 1) << 4;             // causal extent (multiple of 16)
  const int ncols32 = (ncols + 31) & ~31;        // padded to 32 for PV K-blocks

  float* outO = out;                              // [B][L][D]
  float* attn = out + (size_t)B_SZ * L_SZ * D_SZ; // [B][L][L]

  // ---- phase 0: stage Q (fp32->f16); zero fully-masked attn columns -------
  {
    const float4* qsrc = (const float4*)(qg + ((size_t)b * L_SZ + i0) * D_SZ);
    for (int t = tid; t < 16 * D_SZ / 4; t += 256) {
      float4 f = qsrc[t];
      _Float16* dst = Qs + t * 4;
      dst[0] = (_Float16)f.x; dst[1] = (_Float16)f.y;
      dst[2] = (_Float16)f.z; dst[3] = (_Float16)f.w;
    }
    const int zc = L_SZ - ncols;
    if (zc > 0) {
      const int zc4 = zc >> 2;
      const v4f z4 = {0.f, 0.f, 0.f, 0.f};
      for (int t = tid; t < 16 * zc4; t += 256) {
        int r = t / zc4, c4 = t % zc4;
        v4f* p = (v4f*)(attn + ((size_t)(b * L_SZ + i0 + r)) * L_SZ + ncols) + c4;
        __builtin_nontemporal_store(z4, p);   // streaming: don't pollute L2
      }
    }
  }
  __syncthreads();

  // ---- phase 1: logits = (Q.K^T + skew(Q.E^T)) / 8, masked, into LDS ------
  for (int jt = w; jt <= it; jt += 8) {
    const int j0 = jt << 4;

    // async-stage K tile (16 keys x 64 dims, fp32, contiguous 4KB)
    {
      const float* ksrc = kg + ((size_t)b * L_SZ + j0) * D_SZ;
      for (int t = lane; t < 256; t += 32)
        async_b128(kst_off + t * 16, ksrc + t * 4);
    }
    // async-stage E band: rows base-15 .. base+16 (clamped; OOB rows masked)
    const int base = j0 + (L_SZ - 1) - i0;
    for (int t = lane; t < 512; t += 32) {
      int n = t >> 4, c4 = t & 15;
      int er = base - 15 + n;
      er = er < 0 ? 0 : (er > L_SZ - 1 ? L_SZ - 1 : er);
      async_b128(est_off + t * 16, eg + ((size_t)b * L_SZ + er) * D_SZ + c4 * 4);
    }
    wait_async0();   // drain ASYNCcnt: K + E band resident in LDS

    // QK^T tile
    v8f acc = {};
#pragma unroll
    for (int kk = 0; kk < 64; kk += 32) {
      v16h a  = load_a16(Qs, ml, kk, h);
      v16h bb = load_b_f32row(Kst, ml, kk, h);
      acc = WMMA_F16(a, bb, acc);
    }
    // band GEMM: T(16x32) = Q @ E_band^T
    v8f t0 = {}, t1 = {};
#pragma unroll
    for (int kk = 0; kk < 64; kk += 32) {
      v16h a = load_a16(Qs, ml, kk, h);
      t0 = WMMA_F16(a, load_b_f32row(Est, ml,      kk, h), t0);
      t1 = WMMA_F16(a, load_b_f32row(Est, ml + 16, kk, h), t1);
    }
    // spill T, then combine with diagonal gather T[r, c-r+15]
#pragma unroll
    for (int vv = 0; vv < 8; ++vv) {
      int r = vv + 8 * h;
      tsc[r * 32 + ml]      = t0[vv];
      tsc[r * 32 + 16 + ml] = t1[vv];
    }
#pragma unroll
    for (int vv = 0; vv < 8; ++vv) {
      int r = vv + 8 * h;
      int i = i0 + r, j = j0 + ml;
      float val = (j <= i) ? (acc[vv] + tsc[r * 32 + (ml - r + 15)]) * TEMP_INV
                           : NEG_BIG;
      Plds[r * L_SZ + j] = val;
    }
  }
  __syncthreads();

  // ---- phase 2: softmax statistics over the LDS stripe --------------------
  const int rr2 = tid >> 4, ss2 = tid & 15;   // 16 threads per row
  {
    float pm = -3.0e38f;
    for (int c = ss2; c < ncols; c += 16) pm = fmaxf(pm, Plds[rr2 * L_SZ + c]);
    red[rr2 * 16 + ss2] = pm;
  }
  __syncthreads();
  if (tid < 16) {
    float m = red[tid * 16];
    for (int s = 1; s < 16; ++s) m = fmaxf(m, red[tid * 16 + s]);
    mrow[tid] = m;
  }
  __syncthreads();
  {
    const float m = mrow[rr2];
    const int iAbs = i0 + rr2;
    float ps = 0.f;
    for (int c = ss2; c < ncols32; c += 16) {
      float x = Plds[rr2 * L_SZ + c];
      float p = (c <= iAbs) ? __expf(x - m) : 0.0f;   // masked -> exact 0
      Plds[rr2 * L_SZ + c] = p;
      ps += p;
    }
    red[rr2 * 16 + ss2] = ps;
  }
  __syncthreads();
  if (tid < 16) {
    float d = 0.f;
    for (int s = 0; s < 16; ++s) d += red[tid * 16 + s];
    dinv[tid] = 1.0f / d;   // each row has >= 1 unmasked element (exp(0)=1)
  }
  __syncthreads();

  // ---- phase 3a: write normalized attn (once, non-temporal) + PV WMMA -----
  {
    const int cpr = ncols32 >> 2;
    for (int t = tid; t < 16 * cpr; t += 256) {
      int r = t / cpr, c4 = t % cpr;
      v4f p4 = ((const v4f*)(Plds + r * L_SZ))[c4];
      float di = dinv[r];
      p4 *= di;
      v4f* dst = (v4f*)(attn + ((size_t)(b * L_SZ + i0 + r)) * L_SZ) + c4;
      __builtin_nontemporal_store(p4, dst);
    }
  }
  v8f o0 = {}, o1 = {}, o2 = {}, o3 = {};
  for (int jb = w * 32; jb < ncols32; jb += 256) {
    // async-stage V tile (32 rows x 64 dims fp32, contiguous 8KB) into Est
    {
      const float* vsrc = vg + ((size_t)b * L_SZ + jb) * D_SZ;
      for (int t = lane; t < 512; t += 32)
        async_b128(est_off + t * 16, vsrc + t * 4);
    }
    wait_async0();
    v16h a = load_a_f32(Plds, ml, jb, h);   // unnormalized probabilities, K=32
    o0 = WMMA_F16(a, load_b_vcol(Est,  0 + ml, h), o0);
    o1 = WMMA_F16(a, load_b_vcol(Est, 16 + ml, h), o1);
    o2 = WMMA_F16(a, load_b_vcol(Est, 32 + ml, h), o2);
    o3 = WMMA_F16(a, load_b_vcol(Est, 48 + ml, h), o3);
  }
  __syncthreads();

  // ---- phase 3b/3c: reduce O partials across waves (reuse logits LDS) -----
  float* ored = Plds;   // [8 waves][16 rows][64 dims] = 32KB
#pragma unroll
  for (int vv = 0; vv < 8; ++vv) {
    int r = vv + 8 * h;
    ored[w * 1024 + r * 64 +  0 + ml] = o0[vv];
    ored[w * 1024 + r * 64 + 16 + ml] = o1[vv];
    ored[w * 1024 + r * 64 + 32 + ml] = o2[vv];
    ored[w * 1024 + r * 64 + 48 + ml] = o3[vv];
  }
  __syncthreads();
  for (int t = tid; t < 1024; t += 256) {
    int r = t >> 6, dd = t & 63;
    float s = 0.f;
#pragma unroll
    for (int ww = 0; ww < 8; ++ww) s += ored[ww * 1024 + t];
    outO[((size_t)(b * L_SZ + i0 + r)) * D_SZ + dd] = s * dinv[r];
  }
}

extern "C" void kernel_launch(void* const* d_in, const int* in_sizes, int n_in,
                              void* d_out, int out_size, void* d_ws, size_t ws_size,
                              hipStream_t stream) {
  const float* q = (const float*)d_in[0];
  const float* k = (const float*)d_in[1];
  const float* v = (const float*)d_in[2];
  const float* e = (const float*)d_in[3];
  // d_in[4] is the causal mask; it is analytic (j>i) and recomputed in-kernel.
  float* out = (float*)d_out;

  dim3 grid(B_SZ * TILES);   // 2048 workgroups (one per 16-row stripe)
  dim3 block(256);           // 8 wave32s
  hipLaunchKernelGGL(relattn_fused_kernel, grid, block, SMEM_BYTES, stream,
                     q, k, v, e, out);
}
